// KDECrossEntropyLoss_38646115730022
// MI455X (gfx1250) — compile-verified, matrix operand order checked by reading, not run on hardware
//
#include <hip/hip_runtime.h>

typedef __attribute__((ext_vector_type(16))) _Float16 v16h;
typedef __attribute__((ext_vector_type(8)))  _Float16 v8h;
typedef __attribute__((ext_vector_type(8)))  float    v8f;

static constexpr int D      = 256;   // embedding dim (K of the GEMM)
static constexpr int BM     = 128;   // pred rows per block
static constexpr int BN     = 128;   // supp rows per chunk
static constexpr int LDT    = D + 8; // padded LDS row stride (halves): 528B, 16B aligned, conflict-free
static constexpr int NSPLIT = 8;     // grid.y split over N

// ---- CDNA5 async global->LDS (ASYNCcnt-tracked), inline asm for portability ----
__device__ __forceinline__ void async_b128(unsigned lds_off, const void* gaddr) {
  asm volatile("global_load_async_to_lds_b128 %0, %1, off"
               :: "v"(lds_off), "v"(gaddr) : "memory");
}
__device__ __forceinline__ void wait_async0() {
  asm volatile("s_wait_asynccnt 0x0" ::: "memory");
}

// ---------------- init: zero accumulators in workspace ----------------
__global__ __launch_bounds__(256)
void kde_init(float* gnum, float* gden, float* gsum, unsigned* gcnt, int M) {
  int i = blockIdx.x * 256 + threadIdx.x;
  if (i < M) { gnum[i] = 0.f; gden[i] = 0.f; }
  if (i == 0) { *gsum = 0.f; *gcnt = 0u; }
}

// ---------------- prep: fp32 -> f16 + row squared norms ----------------
__global__ __launch_bounds__(64)
void kde_prep(const float* __restrict__ pred, const float* __restrict__ supp,
              _Float16* __restrict__ predh, _Float16* __restrict__ supph,
              float* __restrict__ pn2, float* __restrict__ sn2, int M) {
  int row = blockIdx.x;
  const float* src; _Float16* dst; float* nrm;
  if (row < M) { src = pred + (size_t)row * D; dst = predh + (size_t)row * D; nrm = pn2 + row; }
  else { int r = row - M; src = supp + (size_t)r * D; dst = supph + (size_t)r * D; nrm = sn2 + r; }
  int t = threadIdx.x;                      // 64 threads * 4 floats = 256
  float4 x = ((const float4*)src)[t];
  float s = x.x*x.x + x.y*x.y + x.z*x.z + x.w*x.w;
  dst[t*4+0] = (_Float16)x.x; dst[t*4+1] = (_Float16)x.y;
  dst[t*4+2] = (_Float16)x.z; dst[t*4+3] = (_Float16)x.w;
  #pragma unroll
  for (int m = 1; m < 32; m <<= 1) s += __shfl_xor(s, m, 32);
  __shared__ float ws[2];
  if ((t & 31) == 0) ws[t >> 5] = s;
  __syncthreads();
  if (t == 0) *nrm = ws[0] + ws[1];
}

// ---------------- main fused GEMM + exp + masked row-sums ----------------
__global__ __launch_bounds__(256)
void kde_main(const _Float16* __restrict__ predh, const _Float16* __restrict__ supph,
              const float* __restrict__ pn2, const float* __restrict__ sn2,
              const int* __restrict__ starget, const int* __restrict__ ptarget,
              float* __restrict__ gnum, float* __restrict__ gden, int N) {
  __shared__ __align__(16) _Float16 As[BM][LDT];
  __shared__ __align__(16) _Float16 Bs[2][BN][LDT];
  __shared__ float pn2s[BM]; __shared__ int ptgt[BM];
  __shared__ float dden[BM]; __shared__ float dnum[BM];

  const int tid  = threadIdx.x;
  const int lane = tid & 31, wave = tid >> 5;
  const int l16  = lane & 15, half = lane >> 4;
  const int wm   = wave & 3;        // M quadrant: rows wm*32 .. +31
  const int wn   = wave >> 2;       // N half:   cols wn*64 .. +63
  const int Mbase = blockIdx.x * BM;
  const int chunks_per = (N / BN) / gridDim.y;
  const int c0 = blockIdx.y * chunks_per;

  // prologue: async-load A tile + B chunk 0 straight into LDS (no VGPR staging)
  for (int idx = tid; idx < BM * (D / 8); idx += 256) {
    int r = idx >> 5, g = idx & 31;
    async_b128((unsigned)(uintptr_t)&As[r][g * 8],
               predh + (size_t)(Mbase + r) * D + g * 8);
  }
  {
    const int Nbase = c0 * BN;
    for (int idx = tid; idx < BN * (D / 8); idx += 256) {
      int r = idx >> 5, g = idx & 31;
      async_b128((unsigned)(uintptr_t)&Bs[0][r][g * 8],
                 supph + (size_t)(Nbase + r) * D + g * 8);
    }
  }
  for (int idx = tid; idx < BM; idx += 256) {
    pn2s[idx] = pn2[Mbase + idx];
    ptgt[idx] = ptarget[Mbase + idx];
    dden[idx] = 0.f; dnum[idx] = 0.f;
  }
  wait_async0();
  __syncthreads();

  // per-lane row metadata for this wave's two 16-row M subtiles
  float pr[2][8]; int tg[2][8];
  #pragma unroll
  for (int mt = 0; mt < 2; ++mt) {
    int rb = wm * 32 + mt * 16 + half * 8;
    #pragma unroll
    for (int v = 0; v < 8; ++v) { pr[mt][v] = pn2s[rb + v]; tg[mt][v] = ptgt[rb + v]; }
  }

  for (int ch = 0; ch < chunks_per; ++ch) {
    const int cur = ch & 1;
    if (ch > 0) {              // buffer `cur` was prefetched last iteration
      wait_async0();           // my share landed
      __syncthreads();         // everyone's share landed; everyone done reading cur^1
    }
    if (ch + 1 < chunks_per) { // prefetch next chunk into the other buffer
      const int Nb2 = (c0 + ch + 1) * BN;
      for (int idx = tid; idx < BN * (D / 8); idx += 256) {
        int r = idx >> 5, g = idx & 31;
        async_b128((unsigned)(uintptr_t)&Bs[cur ^ 1][r][g * 8],
                   supph + (size_t)(Nb2 + r) * D + g * 8);
      }
    }
    const int Nbase = (c0 + ch) * BN;

    v8f acc[2][4];
    #pragma unroll
    for (int mt = 0; mt < 2; ++mt)
      #pragma unroll
      for (int nt = 0; nt < 4; ++nt)
        #pragma unroll
        for (int v = 0; v < 8; ++v) acc[mt][nt][v] = 0.f;

    for (int k0 = 0; k0 < D; k0 += 32) {
      // A frags: lane L%16 = M row in subtile; halves [0..7]=k0+half*8.., [8..15]=+16
      v16h af[2];
      #pragma unroll
      for (int mt = 0; mt < 2; ++mt) {
        const _Float16* p = &As[wm * 32 + mt * 16 + l16][k0 + half * 8];
        v8h lo = *(const v8h*)p;
        v8h hi = *(const v8h*)(p + 16);
        af[mt] = __builtin_shufflevector(lo, hi, 0,1,2,3,4,5,6,7,8,9,10,11,12,13,14,15);
      }
      v16h bf[4];
      #pragma unroll
      for (int nt = 0; nt < 4; ++nt) {
        const _Float16* p = &Bs[cur][wn * 64 + nt * 16 + l16][k0 + half * 8];
        v8h lo = *(const v8h*)p;
        v8h hi = *(const v8h*)(p + 16);
        bf[nt] = __builtin_shufflevector(lo, hi, 0,1,2,3,4,5,6,7,8,9,10,11,12,13,14,15);
      }
      #pragma unroll
      for (int mt = 0; mt < 2; ++mt)
        #pragma unroll
        for (int nt = 0; nt < 4; ++nt)
          acc[mt][nt] = __builtin_amdgcn_wmma_f32_16x16x32_f16(
              false, af[mt], false, bf[nt], (short)0, acc[mt][nt], false, false);
    }

    // epilogue: exp kernel + class-masked row sums (meta straight from global/L2)
    #pragma unroll
    for (int mt = 0; mt < 2; ++mt) {
      float ds[8], ns[8];
      #pragma unroll
      for (int v = 0; v < 8; ++v) { ds[v] = 0.f; ns[v] = 0.f; }
      #pragma unroll
      for (int nt = 0; nt < 4; ++nt) {
        int col = wn * 64 + nt * 16 + l16;
        float s2 = sn2[Nbase + col];
        int  cls = starget[Nbase + col];
        #pragma unroll
        for (int v = 0; v < 8; ++v) {
          float sq = pr[mt][v] + s2 - 2.0f * acc[mt][nt][v];
          sq = fmaxf(sq, 0.f);
          float e = __expf(sq * (-1.0f / 256.0f));
          ds[v] += e;
          if (cls == tg[mt][v]) ns[v] += e;
        }
      }
      #pragma unroll
      for (int v = 0; v < 8; ++v) {
        float dsum = ds[v], nsum = ns[v];
        #pragma unroll
        for (int m = 1; m <= 8; m <<= 1) {
          dsum += __shfl_xor(dsum, m, 32);
          nsum += __shfl_xor(nsum, m, 32);
        }
        if (l16 == 0) {
          int row = wm * 32 + mt * 16 + half * 8 + v;
          atomicAdd(&dden[row], dsum);
          atomicAdd(&dnum[row], nsum);
        }
      }
    }
  }

  __syncthreads();
  for (int idx = tid; idx < BM; idx += 256) {
    atomicAdd(&gden[Mbase + idx], dden[idx]);
    atomicAdd(&gnum[Mbase + idx], dnum[idx]);
  }
}

// ---------------- finalize: NLL + mean ----------------
__global__ __launch_bounds__(256)
void kde_fin1(const float* __restrict__ gnum, const float* __restrict__ gden,
              const int* __restrict__ target, float* gsum, unsigned* gcnt, int M) {
  int m = blockIdx.x * 256 + threadIdx.x;
  if (m >= M) return;
  int t = target[m];
  if (t == -100) return;                    // ignore_index
  float den = fmaxf(gden[m], 1e-10f);
  float p = fmaxf(gnum[m] / den, 1e-10f);
  atomicAdd(gsum, -__logf(p));
  atomicAdd(gcnt, 1u);
}

__global__ void kde_fin2(const float* gsum, const unsigned* gcnt, float* out) {
  unsigned c = *gcnt; if (c == 0u) c = 1u;
  out[0] = *gsum / (float)c;
}

extern "C" void kernel_launch(void* const* d_in, const int* in_sizes, int n_in,
                              void* d_out, int out_size, void* d_ws, size_t ws_size,
                              hipStream_t stream) {
  const float* supp    = (const float*)d_in[0];   // (N, D)
  const float* pred    = (const float*)d_in[1];   // (M, D)
  const int*   starget = (const int*)d_in[2];     // (N,)
  const int*   ptarget = (const int*)d_in[3];     // (M,)
  const int N = in_sizes[0] / D;
  const int M = in_sizes[1] / D;

  char* ws = (char*)d_ws;
  _Float16* predh = (_Float16*)ws; ws += (size_t)M * D * 2;
  _Float16* supph = (_Float16*)ws; ws += (size_t)N * D * 2;
  float* pn2  = (float*)ws; ws += (size_t)M * 4;
  float* sn2  = (float*)ws; ws += (size_t)N * 4;
  float* gnum = (float*)ws; ws += (size_t)M * 4;
  float* gden = (float*)ws; ws += (size_t)M * 4;
  float*    gsum = (float*)ws;
  unsigned* gcnt = (unsigned*)(ws + 4);
  float* out = (float*)d_out;

  kde_init<<<(M + 255) / 256, 256, 0, stream>>>(gnum, gden, gsum, gcnt, M);
  kde_prep<<<M + N, 64, 0, stream>>>(pred, supp, predh, supph, pn2, sn2, M);
  kde_main<<<dim3(M / BM, NSPLIT), 256, 0, stream>>>(predh, supph, pn2, sn2,
                                                     starget, ptarget, gnum, gden, N);
  kde_fin1<<<(M + 255) / 256, 256, 0, stream>>>(gnum, gden, ptarget, gsum, gcnt, M);
  kde_fin2<<<1, 1, 0, stream>>>(gsum, gcnt, out);
}
